// MambaMIL_58119497450247
// MI455X (gfx1250) — compile-verified, matrix operand order checked by reading, not run on hardware
//
#include <hip/hip_runtime.h>

// ---------------------------------------------------------------------------
// MambaMIL forward for MI455X (gfx1250, wave32, WMMA).
// GEMMs: operands pre-converted to bf16 once, then v_wmma_f32_16x16x32_bf16
// with 2x2 register blocking per wave, double-buffered LDS (1 barrier/step),
// b128 LDS fragment loads. Selective scan parallelized over (channel,state).
// ---------------------------------------------------------------------------

#define L_SEQ   4096
#define IN_DIM  1024
#define D_MODEL 512
#define D_INNER 1024
#define D_STATE 16
#define D_CONV  4
#define DT_RANK 32
#define N_LAYERS 2

#define ACT_NONE 0
#define ACT_RELU 1
#define ACT_TANH 2
#define ACT_SOFTPLUS 3

typedef __attribute__((ext_vector_type(16))) __bf16          bf16x16;
typedef __attribute__((ext_vector_type(16))) unsigned short  u16x16;
typedef __attribute__((ext_vector_type(8)))  float           f32x8;

__device__ __forceinline__ unsigned short f2bf(float f) {
  unsigned u = __float_as_uint(f);
  unsigned rnd = 0x7fffu + ((u >> 16) & 1u);   // round-to-nearest-even
  return (unsigned short)((u + rnd) >> 16);
}

__device__ __forceinline__ float softplus_f(float x) {
  return (x > 20.f) ? x : log1pf(__expf(x));
}
__device__ __forceinline__ float silu_f(float x) {
  return x / (1.f + __expf(-x));
}

// f32 -> packed bf16 conversion (4 elements / thread)
__global__ void cvt_bf16_k(const float* __restrict__ src,
                           unsigned short* __restrict__ dst, size_t n4)
{
  const size_t i = (size_t)blockIdx.x * blockDim.x + threadIdx.x;
  if (i >= n4) return;
  const float4 v = reinterpret_cast<const float4*>(src)[i];
  ushort4 o;
  o.x = f2bf(v.x); o.y = f2bf(v.y); o.z = f2bf(v.z); o.w = f2bf(v.w);
  reinterpret_cast<ushort4*>(dst)[i] = o;
}

// ---------------------------------------------------------------------------
// Bf16 GEMM: C[M x N] = act(A16[M x K](lda) @ W16[N x K](ldw)^T + bias)
// Block: 128 threads (4 waves), tile 64(M) x 64(N), K stepped by 32.
// Waves 2(M) x 2(N); each wave = 2x2 WMMA tiles -> 4 v_wmma per k-step.
// Double-buffered LDS: prefetch next tile in registers during compute,
// single barrier per k-step. Fragments = two ds_load_b128 each.
// ---------------------------------------------------------------------------
__global__ __launch_bounds__(128) void gemm_wmma_k(
    const unsigned short* __restrict__ A16, int lda,
    const unsigned short* __restrict__ W16, int ldw,
    const float* __restrict__ bias,
    float* __restrict__ C, int ldc,
    int M, int N, int K, int act)
{
  // rows padded to 40 ushorts (80 B): keeps 16B alignment, staggers banks
  __shared__ unsigned short As[2][64][40];
  __shared__ unsigned short Bs[2][64][40];

  const int tid  = threadIdx.x;
  const int m0   = blockIdx.x * 64;
  const int n0   = blockIdx.y * 64;
  const int wave = tid >> 5;
  const int lane = tid & 31;
  const int mt   = wave >> 1;      // 0..1  (32-row half)
  const int nt   = wave & 1;       // 0..1  (32-col half)
  const int hf   = lane >> 4;      // 0..1
  const int idx  = lane & 15;      // 0..15

  f32x8 acc[2][2];
#pragma unroll
  for (int i = 0; i < 2; ++i)
#pragma unroll
    for (int j = 0; j < 2; ++j)
      acc[i][j] = (f32x8){0.f, 0.f, 0.f, 0.f, 0.f, 0.f, 0.f, 0.f};

  // staging: thread t owns row sr = t>>1, 16-ushort half sh = (t&1)*16
  const int sr = tid >> 1;
  const int sh = (tid & 1) << 4;
  const unsigned short* aSrc = A16 + (size_t)(m0 + sr) * lda + sh;
  const unsigned short* wSrc = W16 + (size_t)(n0 + sr) * ldw + sh;

  // prologue: stage k-tile 0 into buffer 0
  {
    const uint4 a0 = *reinterpret_cast<const uint4*>(aSrc);
    const uint4 a1 = *reinterpret_cast<const uint4*>(aSrc + 8);
    const uint4 b0 = *reinterpret_cast<const uint4*>(wSrc);
    const uint4 b1 = *reinterpret_cast<const uint4*>(wSrc + 8);
    *reinterpret_cast<uint4*>(&As[0][sr][sh])     = a0;
    *reinterpret_cast<uint4*>(&As[0][sr][sh + 8]) = a1;
    *reinterpret_cast<uint4*>(&Bs[0][sr][sh])     = b0;
    *reinterpret_cast<uint4*>(&Bs[0][sr][sh + 8]) = b1;
  }
  __syncthreads();

  const int nsteps = K >> 5;
  int cur = 0;
  for (int s = 0; s < nsteps; ++s) {
    // prefetch next k-tile into registers (overlaps with WMMAs below)
    uint4 pa0, pa1, pb0, pb1;
    const bool hasNext = (s + 1 < nsteps);
    if (hasNext) {
      const int k1 = (s + 1) << 5;
      pa0 = *reinterpret_cast<const uint4*>(aSrc + k1);
      pa1 = *reinterpret_cast<const uint4*>(aSrc + k1 + 8);
      pb0 = *reinterpret_cast<const uint4*>(wSrc + k1);
      pb1 = *reinterpret_cast<const uint4*>(wSrc + k1 + 8);
    }

    // fragments: lane needs bf16 runs k=[hf*8, hf*8+8) and [16+hf*8, ...+8)
    // -> exactly two 16B LDS loads per fragment
    bf16x16 afr[2], bfr[2];
#pragma unroll
    for (int i = 0; i < 2; ++i) {
      union { uint4 q[2]; u16x16 v; } ua, ub;
      const unsigned short* ar = &As[cur][mt * 32 + i * 16 + idx][hf * 8];
      const unsigned short* br = &Bs[cur][nt * 32 + i * 16 + idx][hf * 8];
      ua.q[0] = *reinterpret_cast<const uint4*>(ar);
      ua.q[1] = *reinterpret_cast<const uint4*>(ar + 16);
      ub.q[0] = *reinterpret_cast<const uint4*>(br);
      ub.q[1] = *reinterpret_cast<const uint4*>(br + 16);
      afr[i] = __builtin_bit_cast(bf16x16, ua.v);
      bfr[i] = __builtin_bit_cast(bf16x16, ub.v);
    }

    // 4 WMMAs: full fragment reuse, distinct accumulators (no D->A/B hazard)
#pragma unroll
    for (int i = 0; i < 2; ++i)
#pragma unroll
      for (int j = 0; j < 2; ++j)
        acc[i][j] = __builtin_amdgcn_wmma_f32_16x16x32_bf16(
            false, afr[i], false, bfr[j], (short)0, acc[i][j], false, false);

    if (hasNext) {
      const int nxt = cur ^ 1;
      *reinterpret_cast<uint4*>(&As[nxt][sr][sh])     = pa0;
      *reinterpret_cast<uint4*>(&As[nxt][sr][sh + 8]) = pa1;
      *reinterpret_cast<uint4*>(&Bs[nxt][sr][sh])     = pb0;
      *reinterpret_cast<uint4*>(&Bs[nxt][sr][sh + 8]) = pb1;
      __syncthreads();
      cur = nxt;
    }
  }

  // ---- epilogue: C/D layout: lane -> col n=idx; vgpr r -> row hf*8 + r ----
#pragma unroll
  for (int i = 0; i < 2; ++i) {
#pragma unroll
    for (int j = 0; j < 2; ++j) {
      const int col = n0 + nt * 32 + j * 16 + idx;
      const int row_base = m0 + mt * 32 + i * 16 + hf * 8;
      const float bv = bias ? bias[col] : 0.f;
#pragma unroll
      for (int r = 0; r < 8; ++r) {
        float v = acc[i][j][r] + bv;
        if (act == ACT_RELU)          v = fmaxf(v, 0.f);
        else if (act == ACT_TANH)     v = tanhf(v);
        else if (act == ACT_SOFTPLUS) v = softplus_f(v);
        C[(size_t)(row_base + r) * ldc + col] = v;
      }
    }
  }
}

// ---------------------------------------------------------------------------
// LayerNorm: one wave per row (D = 512), 8 rows per block.
// ---------------------------------------------------------------------------
__global__ __launch_bounds__(256) void layernorm_k(
    const float* __restrict__ X, const float* __restrict__ w,
    const float* __restrict__ b, float* __restrict__ Y, int D)
{
  const int row  = blockIdx.x * 8 + (threadIdx.x >> 5);
  const int lane = threadIdx.x & 31;
  const float* x = X + (size_t)row * D;

  float s = 0.f;
  for (int i = lane; i < D; i += 32) s += x[i];
  for (int o = 16; o > 0; o >>= 1) s += __shfl_xor(s, o, 32);
  const float mu = s / (float)D;

  float v = 0.f;
  for (int i = lane; i < D; i += 32) { float d = x[i] - mu; v += d * d; }
  for (int o = 16; o > 0; o >>= 1) v += __shfl_xor(v, o, 32);
  const float rs = rsqrtf(v / (float)D + 1e-5f);

  float* y = Y + (size_t)row * D;
  for (int i = lane; i < D; i += 32) y[i] = (x[i] - mu) * rs * w[i] + b[i];
}

// ---------------------------------------------------------------------------
// perm[j] for perm = concat([arange(i, L, rate) for i in range(rate)])
// ---------------------------------------------------------------------------
__global__ void build_perm_k(const int* __restrict__ rate_p,
                             int* __restrict__ perm, int L)
{
  const int j = blockIdx.x * blockDim.x + threadIdx.x;
  if (j >= L) return;
  const int rate = rate_p[0];
  int off = 0;
  for (int i = 0; i < rate; ++i) {
    const int cnt = (L - i + rate - 1) / rate;
    if (j < off + cnt) { perm[j] = i + (j - off) * rate; return; }
    off += cnt;
  }
}

// Gather + split: xp[j,c] = xz[perm[j], c]; zp[j,c] = xz[perm[j], DI + c]
__global__ void gather_split_k(const float* __restrict__ xz,
                               const int* __restrict__ perm,
                               float* __restrict__ xp, float* __restrict__ zp,
                               int L, int DI)
{
  const size_t i = (size_t)blockIdx.x * blockDim.x + threadIdx.x;
  if (i >= (size_t)L * DI) return;
  const int j = (int)(i / DI), c = (int)(i % DI);
  const size_t src = (size_t)perm[j] * (2 * DI);
  xp[i] = xz[src + c];
  zp[i] = xz[src + DI + c];
}

// Causal depthwise conv (K=4) + SiLU
__global__ void conv_silu_k(const float* __restrict__ xp,
                            const float* __restrict__ cw,
                            const float* __restrict__ cb,
                            float* __restrict__ xc, int L, int DI)
{
  const size_t i = (size_t)blockIdx.x * blockDim.x + threadIdx.x;
  if (i >= (size_t)L * DI) return;
  const int t = (int)(i / DI), c = (int)(i % DI);
  float acc = cb[c];
#pragma unroll
  for (int k = 0; k < D_CONV; ++k) {
    const int tt = t - (D_CONV - 1) + k;
    const float xv = (tt >= 0) ? xp[(size_t)tt * DI + c] : 0.f;
    acc += xv * cw[c * D_CONV + k];
  }
  xc[i] = silu_f(acc);
}

// ---------------------------------------------------------------------------
// Selective scan: lane = (channel pair-half, state). 2 channels / wave,
// width-16 shuffle reduction for h . C each step. y includes D-skip & silu(z).
// ---------------------------------------------------------------------------
__global__ __launch_bounds__(256) void scan_k(
    const float* __restrict__ xc, const float* __restrict__ dt,
    const float* __restrict__ dbl, const float* __restrict__ zp,
    const float* __restrict__ A_log, const float* __restrict__ Dsk,
    float* __restrict__ y, int L, int DI)
{
  const int lane = threadIdx.x & 31;
  const int wave = blockIdx.x * (blockDim.x >> 5) + (threadIdx.x >> 5);
  const int c = wave * 2 + (lane >> 4);
  const int s = lane & 15;

  const float As = -__expf(A_log[c * D_STATE + s]);
  const float Dc = Dsk[c];
  float h = 0.f;

  for (int t = 0; t < L; ++t) {
    const float dtv = dt[(size_t)t * DI + c];
    const float u   = xc[(size_t)t * DI + c];
    const float Bv  = dbl[(size_t)t * 64 + DT_RANK + s];
    const float Cv  = dbl[(size_t)t * 64 + DT_RANK + D_STATE + s];
    const float dA  = __expf(dtv * As);
    h = dA * h + (dtv * u) * Bv;
    float p = h * Cv;
    for (int o = 8; o > 0; o >>= 1) p += __shfl_xor(p, o, 16);
    if (s == 0) {
      const float z = zp[(size_t)t * DI + c];
      y[(size_t)t * DI + c] = (p + u * Dc) * silu_f(z);
    }
  }
}

// Scatter (un-permute): yu[perm[j], c] = y[j, c]
__global__ void scatter_k(const float* __restrict__ y,
                          const int* __restrict__ perm,
                          float* __restrict__ yu, int L, int DI)
{
  const size_t i = (size_t)blockIdx.x * blockDim.x + threadIdx.x;
  if (i >= (size_t)L * DI) return;
  const int j = (int)(i / DI), c = (int)(i % DI);
  yu[(size_t)perm[j] * DI + c] = y[i];
}

__global__ void add_inplace_k(float* __restrict__ h,
                              const float* __restrict__ t, size_t n)
{
  const size_t i = (size_t)blockIdx.x * blockDim.x + threadIdx.x;
  if (i < n) h[i] += t[i];
}

// a[t] = t1[t,:] . w2 + b2     (128-length dot per row)
__global__ void attn_score_k(const float* __restrict__ t1,
                             const float* __restrict__ w2,
                             const float* __restrict__ b2,
                             float* __restrict__ a, int L)
{
  const int t = blockIdx.x * blockDim.x + threadIdx.x;
  if (t >= L) return;
  float s = b2[0];
#pragma unroll 4
  for (int i = 0; i < 128; ++i) s += t1[(size_t)t * 128 + i] * w2[i];
  a[t] = s;
}

// Softmax over L elements, single block of 1024 threads.
__global__ __launch_bounds__(1024) void softmax_k(
    const float* __restrict__ a, float* __restrict__ out, int L)
{
  __shared__ float red[32];
  const int tid = threadIdx.x;

  float m = -1e30f;
  for (int i = tid; i < L; i += 1024) m = fmaxf(m, a[i]);
  for (int o = 16; o > 0; o >>= 1) m = fmaxf(m, __shfl_xor(m, o, 32));
  if ((tid & 31) == 0) red[tid >> 5] = m;
  __syncthreads();
  if (tid < 32) {
    float v = red[tid];
    for (int o = 16; o > 0; o >>= 1) v = fmaxf(v, __shfl_xor(v, o, 32));
    red[tid] = v;
  }
  __syncthreads();
  m = red[0];
  __syncthreads();

  float s = 0.f;
  for (int i = tid; i < L; i += 1024) s += __expf(a[i] - m);
  for (int o = 16; o > 0; o >>= 1) s += __shfl_xor(s, o, 32);
  if ((tid & 31) == 0) red[tid >> 5] = s;
  __syncthreads();
  if (tid < 32) {
    float v = red[tid];
    for (int o = 16; o > 0; o >>= 1) v += __shfl_xor(v, o, 32);
    red[tid] = v;
  }
  __syncthreads();
  const float inv = 1.f / red[0];
  for (int i = tid; i < L; i += 1024) out[i] = __expf(a[i] - m) * inv;
}

// pooled[c] = sum_t Aatt[t] * hn[t, c]   (one block per c)
__global__ __launch_bounds__(256) void pool_k(
    const float* __restrict__ Aatt, const float* __restrict__ hn,
    float* __restrict__ pooled, int L, int D)
{
  __shared__ float red[8];
  const int c = blockIdx.x;
  float s = 0.f;
  for (int t = threadIdx.x; t < L; t += 256) s += Aatt[t] * hn[(size_t)t * D + c];
  for (int o = 16; o > 0; o >>= 1) s += __shfl_xor(s, o, 32);
  if ((threadIdx.x & 31) == 0) red[threadIdx.x >> 5] = s;
  __syncthreads();
  if (threadIdx.x == 0) {
    float tot = 0.f;
    for (int i = 0; i < 8; ++i) tot += red[i];
    pooled[c] = tot;
  }
}

// out[n] = pooled . out2048_w[n,:] + b[n]
__global__ void final_out_k(const float* __restrict__ pooled,
                            const float* __restrict__ w,
                            const float* __restrict__ b,
                            float* __restrict__ out, int N, int D)
{
  const int n = blockIdx.x * blockDim.x + threadIdx.x;
  if (n >= N) return;
  float s = b[n];
#pragma unroll 4
  for (int i = 0; i < 512; ++i) s += pooled[i] * w[(size_t)n * D + i];
  out[n] = s;
}

// ---------------------------------------------------------------------------
extern "C" void kernel_launch(void* const* d_in, const int* in_sizes, int n_in,
                              void* d_out, int out_size, void* d_ws, size_t ws_size,
                              hipStream_t stream) {
  (void)in_sizes; (void)n_in; (void)out_size; (void)ws_size;

  const float* x        = (const float*)d_in[0];
  const int*   rate_p   = (const int*)  d_in[1];
  const float* fc1_w    = (const float*)d_in[2];
  const float* fc1_b    = (const float*)d_in[3];
  const float* ln_w     = (const float*)d_in[4];
  const float* ln_b     = (const float*)d_in[5];
  const float* in_proj  = (const float*)d_in[6];
  const float* conv_w   = (const float*)d_in[7];
  const float* conv_b   = (const float*)d_in[8];
  const float* x_proj   = (const float*)d_in[9];
  const float* dt_pw    = (const float*)d_in[10];
  const float* dt_pb    = (const float*)d_in[11];
  const float* A_log    = (const float*)d_in[12];
  const float* Dsk      = (const float*)d_in[13];
  const float* out_proj = (const float*)d_in[14];
  const float* norm_w   = (const float*)d_in[15];
  const float* norm_b   = (const float*)d_in[16];
  const float* attn_w1  = (const float*)d_in[17];
  const float* attn_b1  = (const float*)d_in[18];
  const float* attn_w2  = (const float*)d_in[19];
  const float* attn_b2  = (const float*)d_in[20];
  const float* o2k_w    = (const float*)d_in[21];
  const float* o2k_b    = (const float*)d_in[22];

  float* out = (float*)d_out;                 // [0,2048) logits, [2048,6144) Aatt

  // ---- workspace carve ----
  char* p = (char*)d_ws;
  auto carve = [&](size_t bytes) {
    void* q = (void*)p;
    p += (bytes + 255) & ~(size_t)255;
    return q;
  };
  int*   perm = (int*)  carve((size_t)L_SEQ * 4);
  float* h    = (float*)carve((size_t)L_SEQ * D_MODEL * 4);
  float* ln   = (float*)carve((size_t)L_SEQ * D_MODEL * 4);
  float* xz   = (float*)carve((size_t)L_SEQ * 2 * D_INNER * 4);  // also hosts y,yu
  float* xp   = (float*)carve((size_t)L_SEQ * D_INNER * 4);
  float* zp   = (float*)carve((size_t)L_SEQ * D_INNER * 4);
  float* xc   = (float*)carve((size_t)L_SEQ * D_INNER * 4);
  float* dbl  = (float*)carve((size_t)L_SEQ * 64 * 4);
  float* dtb  = (float*)carve((size_t)L_SEQ * D_INNER * 4);
  float* tmp  = (float*)carve((size_t)L_SEQ * D_MODEL * 4);
  float* hn   = (float*)carve((size_t)L_SEQ * D_MODEL * 4);
  float* t1   = (float*)carve((size_t)L_SEQ * 128 * 4);
  float* asc  = (float*)carve((size_t)L_SEQ * 4);
  float* pooled = (float*)carve((size_t)D_MODEL * 4);
  unsigned short* abuf = (unsigned short*)carve((size_t)L_SEQ * IN_DIM * 2);     // bf16 A
  unsigned short* wbuf = (unsigned short*)carve((size_t)2 * D_INNER * D_MODEL * 2); // bf16 W
  // alias y/yu into xz (xz is dead after gather_split)
  float* y  = xz;
  float* yu = xz + (size_t)L_SEQ * D_INNER;

  auto gemm = [&](const float* A, int lda, const float* W, int ldw,
                  const float* bias, float* C, int ldc,
                  int M, int N, int K, int act) {
    const size_t na4 = (size_t)M * lda / 4;
    const size_t nw4 = (size_t)N * ldw / 4;
    cvt_bf16_k<<<(int)((na4 + 255) / 256), 256, 0, stream>>>(A, abuf, na4);
    cvt_bf16_k<<<(int)((nw4 + 255) / 256), 256, 0, stream>>>(W, wbuf, nw4);
    dim3 grid(M / 64, N / 64);
    gemm_wmma_k<<<grid, 128, 0, stream>>>(abuf, lda, wbuf, ldw, bias, C, ldc, M, N, K, act);
  };

  const size_t nLD = (size_t)L_SEQ * D_INNER;
  const int eltBlocks = (int)((nLD + 255) / 256);

  // perm table
  build_perm_k<<<(L_SEQ + 255) / 256, 256, 0, stream>>>(rate_p, perm, L_SEQ);

  // h = relu(x @ fc1_w^T + fc1_b)
  gemm(x, IN_DIM, fc1_w, IN_DIM, fc1_b, h, D_MODEL, L_SEQ, D_MODEL, IN_DIM, ACT_RELU);

  for (int l = 0; l < N_LAYERS; ++l) {
    layernorm_k<<<L_SEQ / 8, 256, 0, stream>>>(h, ln_w + l * D_MODEL, ln_b + l * D_MODEL, ln, D_MODEL);

    // xz = ln @ in_proj^T
    gemm(ln, D_MODEL, in_proj + (size_t)l * 2 * D_INNER * D_MODEL, D_MODEL,
         nullptr, xz, 2 * D_INNER, L_SEQ, 2 * D_INNER, D_MODEL, ACT_NONE);

    gather_split_k<<<eltBlocks, 256, 0, stream>>>(xz, perm, xp, zp, L_SEQ, D_INNER);

    conv_silu_k<<<eltBlocks, 256, 0, stream>>>(
        xp, conv_w + (size_t)l * D_INNER * D_CONV, conv_b + (size_t)l * D_INNER,
        xc, L_SEQ, D_INNER);

    // dbl = xc @ x_proj^T   (N = 64: dt_rank | B | C)
    gemm(xc, D_INNER, x_proj + (size_t)l * 64 * D_INNER, D_INNER,
         nullptr, dbl, 64, L_SEQ, 64, D_INNER, ACT_NONE);

    // dt = softplus(dbl[:, :32] @ dt_proj^T + dt_b)
    gemm(dbl, 64, dt_pw + (size_t)l * D_INNER * DT_RANK, DT_RANK,
         dt_pb + (size_t)l * D_INNER, dtb, D_INNER, L_SEQ, D_INNER, DT_RANK, ACT_SOFTPLUS);

    scan_k<<<D_INNER / 16, 256, 0, stream>>>(
        xc, dtb, dbl, zp, A_log + (size_t)l * D_INNER * D_STATE,
        Dsk + (size_t)l * D_INNER, y, L_SEQ, D_INNER);

    scatter_k<<<eltBlocks, 256, 0, stream>>>(y, perm, yu, L_SEQ, D_INNER);

    // tmp = yu @ out_proj^T ; h += tmp
    gemm(yu, D_INNER, out_proj + (size_t)l * D_MODEL * D_INNER, D_INNER,
         nullptr, tmp, D_MODEL, L_SEQ, D_MODEL, D_INNER, ACT_NONE);
    add_inplace_k<<<(int)(((size_t)L_SEQ * D_MODEL + 255) / 256), 256, 0, stream>>>(
        h, tmp, (size_t)L_SEQ * D_MODEL);
  }

  layernorm_k<<<L_SEQ / 8, 256, 0, stream>>>(h, norm_w, norm_b, hn, D_MODEL);

  // t1 = tanh(hn @ attn_w1^T + attn_b1)
  gemm(hn, D_MODEL, attn_w1, D_MODEL, attn_b1, t1, 128, L_SEQ, 128, D_MODEL, ACT_TANH);

  attn_score_k<<<(L_SEQ + 255) / 256, 256, 0, stream>>>(t1, attn_w2, attn_b2, asc, L_SEQ);

  softmax_k<<<1, 1024, 0, stream>>>(asc, out + 2048, L_SEQ);     // Aatt -> d_out[2048:]

  pool_k<<<D_MODEL, 256, 0, stream>>>(out + 2048, hn, pooled, L_SEQ, D_MODEL);

  final_out_k<<<(2048 + 255) / 256, 256, 0, stream>>>(pooled, o2k_w, o2k_b, out, 2048, D_MODEL);
}
